// GNNNet_38714835206889
// MI455X (gfx1250) — matrix operand order repaired and checked by reading, not compiled, not run on hardware
//
#include <hip/hip_runtime.h>
#include <hip/hip_bf16.h>

// ---------------------------------------------------------------------------
// GNN (SAGE-conv stack) for MI455X / gfx1250, wave32 + WMMA.
//
// Roofline: edge aggregation (~1.6 GB L2 traffic/layer, atomics hit the
// 192 MB L2 since each activation buffer is ~102 MB) dominates; the 150 GFLOP
// of GEMMs run on v_wmma_f32_16x16x32_bf16 using the split-bf16 (hi+lo,
// 3-product) scheme for near-fp32 accuracy at matrix-core rate.
// ---------------------------------------------------------------------------

#define N_NODES 100000
#define N_EDGES 800000
#define N_GRAPHS 128
#define NPAD 100096   // multiple of 128 (8 waves x 16 rows) -> no EXEC guards
#define LD 256        // fixed row stride of all activation buffers

typedef __attribute__((ext_vector_type(16))) __bf16 v16bf;
typedef __attribute__((ext_vector_type(8)))  float  v8f;

__device__ __forceinline__ float elu1(float v) {
  return v > 0.f ? v : (expf(v) - 1.f);
}

// ----------------------------- utility kernels -----------------------------

__global__ void k_zero(float* __restrict__ p, int n) {
  int i = blockIdx.x * blockDim.x + threadIdx.x;
  if (i < n) p[i] = 0.f;
}

// hA[n][0..12] = x[n], cols 13..31 zeroed (layer-1 K padded to 32), pad rows zero
__global__ void k_copy_x(const float* __restrict__ x, float* __restrict__ h) {
  int i = blockIdx.x * blockDim.x + threadIdx.x;   // NPAD*32 threads
  int n = i >> 5, c = i & 31;
  if (n >= NPAD) return;
  float v = 0.f;
  if (c < 13 && n < N_NODES) v = x[n * 13 + c];
  h[(long)n * LD + c] = v;
}

__global__ void k_deg(const int* __restrict__ dst, float* __restrict__ deg) {
  int e = blockIdx.x * blockDim.x + threadIdx.x;
  if (e < N_EDGES) atomicAdd(&deg[dst[e]], 1.f);
}

__global__ void k_invdeg(float* __restrict__ deg, int n) {
  int i = blockIdx.x * blockDim.x + threadIdx.x;
  if (i < n) deg[i] = 1.f / fmaxf(deg[i], 1.f);
}

// zero agg[:, 0..Kpad) where Kpad = 1<<ksh
__global__ void k_zero_cols(float* __restrict__ agg, int ksh, int total) {
  int i = blockIdx.x * blockDim.x + threadIdx.x;
  if (i >= total) return;
  int n = i >> ksh;
  int c = i & ((1 << ksh) - 1);
  agg[(long)n * LD + c] = 0.f;
}

// agg[dst] += h[src], float4 chunks; chunks per edge = 1<<csh
__global__ void k_scatter(const float* __restrict__ h, const int* __restrict__ src,
                          const int* __restrict__ dst, float* __restrict__ agg,
                          int csh, long total) {
  long i = (long)blockIdx.x * blockDim.x + threadIdx.x;
  if (i >= total) return;
  int e = (int)(i >> csh);
  int c = ((int)i & ((1 << csh) - 1)) << 2;
  const float4 v = *(const float4*)(h + (long)src[e] * LD + c);
  float* a = agg + (long)dst[e] * LD + c;
  atomicAdd(a + 0, v.x);
  atomicAdd(a + 1, v.y);
  atomicAdd(a + 2, v.z);
  atomicAdd(a + 3, v.w);
}

// Convert f32 weight [K, dout] into hi/lo bf16 arrays stored in wave32
// B-fragment order for v_wmma_*_16x16x32_bf16:
//   fragment (kstep t, n-tile j): element e at lane L is
//     k = t*32 + (L>=16)*8 + (e&7) + (e>=8)*16,  n = j*16 + (L&15)
//   stored flat at ((t*(dout/16)+j)*512 + L*16 + e)
__global__ void k_wconv(const float* __restrict__ W, __bf16* __restrict__ hi,
                        __bf16* __restrict__ lo, int K, int dout, int total) {
  int i = blockIdx.x * blockDim.x + threadIdx.x;
  if (i >= total) return;
  int e = i & 15;
  int L = (i >> 4) & 31;
  int frag = i >> 9;
  int nt = dout >> 4;
  int t = frag / nt;
  int j = frag - t * nt;
  int k = t * 32 + ((L >> 4) << 3) + (e & 7) + ((e >> 3) << 4);
  int n = (j << 4) + (L & 15);
  float f = (k < K) ? W[(long)k * dout + n] : 0.f;
  __bf16 h = (__bf16)f;
  hi[i] = h;
  lo[i] = (__bf16)(f - (float)h);
}

// ------------------------------- WMMA GEMM ---------------------------------

// Build A fragment (16x32 bf16 hi/lo) for one wave from f32 activations.
// Per ISA 7.12.2: lane<16 -> row m0+lane, K = {kk..kk+7, kk+16..kk+23}
//                 lane>=16 -> row m0+lane-16, K shifted by +8.
__device__ __forceinline__ void load_a(const float* __restrict__ p, float s,
                                       v16bf& ah, v16bf& al) {
  float f[16];
#pragma unroll
  for (int q = 0; q < 8; ++q) f[q] = p[q];
#pragma unroll
  for (int q = 0; q < 8; ++q) f[8 + q] = p[16 + q];
#pragma unroll
  for (int q = 0; q < 16; ++q) {
    float v = f[q] * s;
    __bf16 h = (__bf16)v;
    ah[q] = h;
    al[q] = (__bf16)(v - (float)h);
  }
}

// Split-fp32 product: hi*hi + hi*lo + lo*hi (lo*lo ~2^-18 rel, dropped)
__device__ __forceinline__ v8f mm3(v8f c, const v16bf& ah, const v16bf& al,
                                   const __bf16* __restrict__ whi,
                                   const __bf16* __restrict__ wlo, long off) {
  v16bf bh = *(const v16bf*)(whi + off);
  v16bf bl = *(const v16bf*)(wlo + off);
  c = __builtin_amdgcn_wmma_f32_16x16x32_bf16(false, ah, false, bh, (short)0, c, false, false);
  c = __builtin_amdgcn_wmma_f32_16x16x32_bf16(false, ah, false, bl, (short)0, c, false, false);
  c = __builtin_amdgcn_wmma_f32_16x16x32_bf16(false, al, false, bh, (short)0, c, false, false);
  return c;
}

// C/D layout (ISA 7.12.2): elem v at lane L -> row m0 + v + (L>=16)*8, col lane&15
__device__ __forceinline__ void epi(const v8f a, int n, int rbase,
                                    const float* __restrict__ bias,
                                    const float* __restrict__ res,
                                    float* __restrict__ out) {
  float b = bias[n];
#pragma unroll
  for (int v = 0; v < 8; ++v) {
    float val = elu1(a[v] + b);
    if (res) val += res[(long)(rbase + v) * LD + n];
    out[(long)(rbase + v) * LD + n] = val;
  }
}

// out = [res +] elu( (A1 * invdeg_row) @ W1 + bias [+ A2 @ W2] )
// 8 waves/block, each wave -> 16 rows x 64 cols (4 accumulator tiles).
__global__ __launch_bounds__(256) void k_gemm(
    const float* __restrict__ A1, const float* __restrict__ invdeg,
    const __bf16* __restrict__ W1hi, const __bf16* __restrict__ W1lo,
    const float* __restrict__ A2,
    const __bf16* __restrict__ W2hi, const __bf16* __restrict__ W2lo,
    const float* __restrict__ bias, const float* __restrict__ res,
    float* __restrict__ out, int Kpad, int dout) {
  const int lane = threadIdx.x & 31;
  const int wave = threadIdx.x >> 5;
  const int m0 = (blockIdx.x * 8 + wave) * 16;
  const int n_base = blockIdx.y * 64;
  const int row = m0 + (lane & 15);
  const int kofs = (lane < 16) ? 0 : 8;
  const int ntiles = dout >> 4;
  const long fb0 = (long)(n_base >> 4) * 512 + lane * 16;

  v8f c0 = {}, c1 = {}, c2 = {}, c3 = {};
  const float s1 = invdeg ? invdeg[row] : 1.f;

  for (int kk = 0; kk < Kpad; kk += 32) {
    const long fb = (long)(kk >> 5) * (ntiles * 512) + fb0;
    v16bf ah, al;
    load_a(A1 + (long)row * LD + kk + kofs, s1, ah, al);
    c0 = mm3(c0, ah, al, W1hi, W1lo, fb);
    c1 = mm3(c1, ah, al, W1hi, W1lo, fb + 512);
    c2 = mm3(c2, ah, al, W1hi, W1lo, fb + 1024);
    c3 = mm3(c3, ah, al, W1hi, W1lo, fb + 1536);
    if (A2) {
      load_a(A2 + (long)row * LD + kk + kofs, 1.f, ah, al);
      c0 = mm3(c0, ah, al, W2hi, W2lo, fb);
      c1 = mm3(c1, ah, al, W2hi, W2lo, fb + 512);
      c2 = mm3(c2, ah, al, W2hi, W2lo, fb + 1024);
      c3 = mm3(c3, ah, al, W2hi, W2lo, fb + 1536);
    }
  }

  const int colL = lane & 15;
  const int rbase = m0 + ((lane < 16) ? 0 : 8);
  epi(c0, n_base + colL,      rbase, bias, res, out);
  epi(c1, n_base + 16 + colL, rbase, bias, res, out);
  epi(c2, n_base + 32 + colL, rbase, bias, res, out);
  epi(c3, n_base + 48 + colL, rbase, bias, res, out);
}

// ------------------------------ pool + tail --------------------------------

__global__ void k_pool(const float* __restrict__ h, const int* __restrict__ batch,
                       float* __restrict__ pooled, float* __restrict__ cnt) {
  int i = blockIdx.x * blockDim.x + threadIdx.x;  // N_NODES*32
  if (i >= N_NODES * 32) return;
  int n = i >> 5, c = (i & 31) << 2;
  int g = batch[n];
  const float4 v = *(const float4*)(h + (long)n * LD + c);
  float* p = pooled + g * 128 + c;
  atomicAdd(p + 0, v.x);
  atomicAdd(p + 1, v.y);
  atomicAdd(p + 2, v.z);
  atomicAdd(p + 3, v.w);
  if (c == 0) atomicAdd(&cnt[g], 1.f);
}

__global__ void k_tail(const float* __restrict__ pooled, const float* __restrict__ cnt,
                       const float* __restrict__ bg, const float* __restrict__ bb,
                       const float* __restrict__ rm, const float* __restrict__ rv,
                       const float* __restrict__ W3, const float* __restrict__ b3,
                       const float* __restrict__ Wo, const float* __restrict__ bo,
                       float* __restrict__ out) {
  __shared__ float hg[128];
  __shared__ float y[64];
  __shared__ float o[3];
  int g = blockIdx.x, t = threadIdx.x;
  float c = fmaxf(cnt[g], 1.f);
  if (t < 128) {
    float p = pooled[g * 128 + t] / c;
    hg[t] = (p - rm[t]) * (1.f / sqrtf(rv[t] + 1e-5f)) * bg[t] + bb[t];
  }
  __syncthreads();
  if (t < 64) {
    float s = b3[t];
    for (int k = 0; k < 128; ++k) s += hg[k] * W3[k * 64 + t];
    y[t] = elu1(s);
  }
  __syncthreads();
  if (t < 3) {
    float s = bo[t];
    for (int k = 0; k < 64; ++k) s += y[k] * Wo[k * 3 + t];
    o[t] = s;
  }
  __syncthreads();
  if (t == 0) {
    float nrm = sqrtf(o[0] * o[0] + o[1] * o[1] + o[2] * o[2]);
    nrm = fmaxf(nrm, 1e-12f);
    out[g * 3 + 0] = o[0] / nrm;
    out[g * 3 + 1] = o[1] / nrm;
    out[g * 3 + 2] = o[2] / nrm;
  }
}

// -------------------------------- launcher ---------------------------------

extern "C" void kernel_launch(void* const* d_in, const int* in_sizes, int n_in,
                              void* d_out, int out_size, void* d_ws, size_t ws_size,
                              hipStream_t stream) {
  (void)in_sizes; (void)n_in; (void)out_size; (void)ws_size;
  const int*   batch = (const int*)d_in[0];
  const float* x     = (const float*)d_in[1];
  const int*   ei    = (const int*)d_in[2];
  const int*   src   = ei;
  const int*   dst   = ei + N_EDGES;
  // d_in[3] = pseudo (unused by the reference math)
  const float *convWl[6], *convbl[6], *convWr[6];
  for (int l = 0; l < 6; ++l) {
    convWl[l] = (const float*)d_in[4 + l * 3];
    convbl[l] = (const float*)d_in[5 + l * 3];
    convWr[l] = (const float*)d_in[6 + l * 3];
  }
  const float* fc1_W = (const float*)d_in[22]; const float* fc1_b = (const float*)d_in[23];
  const float* fc2_W = (const float*)d_in[24]; const float* fc2_b = (const float*)d_in[25];
  const float* bn_g  = (const float*)d_in[26]; const float* bn_b  = (const float*)d_in[27];
  const float* bn_rm = (const float*)d_in[28]; const float* bn_rv = (const float*)d_in[29];
  const float* fc3_W = (const float*)d_in[30]; const float* fc3_b = (const float*)d_in[31];
  const float* out_W = (const float*)d_in[32]; const float* out_b = (const float*)d_in[33];

  size_t off = 0;
  char* base = (char*)d_ws;
  auto carve = [&](size_t bytes) -> char* {
    char* p = base + off;
    off += (bytes + 255) & ~(size_t)255;
    return p;
  };
  float* hA     = (float*)carve((size_t)NPAD * LD * 4);
  float* hB     = (float*)carve((size_t)NPAD * LD * 4);
  float* agg    = (float*)carve((size_t)NPAD * LD * 4);
  float* deg    = (float*)carve((size_t)NPAD * 4);
  float* pooled = (float*)carve(128 * 128 * 4);
  float* gcnt   = (float*)carve(128 * 4);

  const int Kp[6] = {32, 128, 256, 256, 256, 256};   // padded K
  const int Kr[6] = {13, 128, 256, 256, 256, 256};   // real K
  const int Do[6] = {128, 256, 256, 256, 256, 256};
  const int ksh[6] = {5, 7, 8, 8, 8, 8};             // log2(Kp)
  const int csh[6] = {2, 5, 6, 6, 6, 6};             // log2(float4 chunks/edge)
  __bf16 *Wlhi[6], *Wllo[6], *Wrhi[6], *Wrlo[6];
  for (int l = 0; l < 6; ++l) {
    size_t e = (size_t)Kp[l] * Do[l];
    Wlhi[l] = (__bf16*)carve(e * 2); Wllo[l] = (__bf16*)carve(e * 2);
    Wrhi[l] = (__bf16*)carve(e * 2); Wrlo[l] = (__bf16*)carve(e * 2);
  }
  __bf16* F1hi = (__bf16*)carve(256 * 256 * 2); __bf16* F1lo = (__bf16*)carve(256 * 256 * 2);
  __bf16* F2hi = (__bf16*)carve(256 * 128 * 2); __bf16* F2lo = (__bf16*)carve(256 * 128 * 2);

  const int B = 256;

  // init
  k_copy_x<<<(NPAD * 32 + B - 1) / B, B, 0, stream>>>(x, hA);
  k_zero<<<(NPAD + B - 1) / B, B, 0, stream>>>(deg, NPAD);
  k_zero<<<(128 * 128 + B - 1) / B, B, 0, stream>>>(pooled, 128 * 128);
  k_zero<<<1, B, 0, stream>>>(gcnt, 128);
  k_deg<<<(N_EDGES + B - 1) / B, B, 0, stream>>>(dst, deg);
  k_invdeg<<<(NPAD + B - 1) / B, B, 0, stream>>>(deg, NPAD);

  // weight pre-conversion to split-bf16 fragment layout
  for (int l = 0; l < 6; ++l) {
    int tot = Kp[l] * Do[l];
    k_wconv<<<(tot + B - 1) / B, B, 0, stream>>>(convWl[l], Wlhi[l], Wllo[l], Kr[l], Do[l], tot);
    k_wconv<<<(tot + B - 1) / B, B, 0, stream>>>(convWr[l], Wrhi[l], Wrlo[l], Kr[l], Do[l], tot);
  }
  k_wconv<<<(256 * 256 + B - 1) / B, B, 0, stream>>>(fc1_W, F1hi, F1lo, 256, 256, 256 * 256);
  k_wconv<<<(256 * 128 + B - 1) / B, B, 0, stream>>>(fc2_W, F2hi, F2lo, 256, 128, 256 * 128);

  float* cur = hA;
  float* nxt = hB;
  for (int l = 0; l < 6; ++l) {
    int ztot = NPAD << ksh[l];
    k_zero_cols<<<(ztot + B - 1) / B, B, 0, stream>>>(agg, ksh[l], ztot);
    long stot = (long)N_EDGES << csh[l];
    k_scatter<<<(int)((stot + B - 1) / B), B, 0, stream>>>(cur, src, dst, agg, csh[l], stot);
    dim3 g(NPAD / 128, Do[l] / 64);
    k_gemm<<<g, B, 0, stream>>>(agg, deg, Wlhi[l], Wllo[l], cur, Wrhi[l], Wrlo[l],
                                convbl[l], (l >= 2) ? cur : nullptr, nxt, Kp[l], Do[l]);
    float* t = cur; cur = nxt; nxt = t;
  }
  { // fc1: h = h + elu(h @ W + b)
    dim3 g(NPAD / 128, 4);
    k_gemm<<<g, B, 0, stream>>>(cur, nullptr, F1hi, F1lo, nullptr, nullptr, nullptr,
                                fc1_b, cur, nxt, 256, 256);
    float* t = cur; cur = nxt; nxt = t;
  }
  { // fc2: h = elu(h @ W + b), dout=128
    dim3 g(NPAD / 128, 2);
    k_gemm<<<g, B, 0, stream>>>(cur, nullptr, F2hi, F2lo, nullptr, nullptr, nullptr,
                                fc2_b, nullptr, nxt, 256, 128);
    float* t = cur; cur = nxt; nxt = t;
  }
  k_pool<<<(N_NODES * 32 + B - 1) / B, B, 0, stream>>>(cur, batch, pooled, gcnt);
  k_tail<<<N_GRAPHS, 128, 0, stream>>>(pooled, gcnt, bn_g, bn_b, bn_rm, bn_rv,
                                       fc3_W, fc3_b, out_W, out_b, (float*)d_out);
}
